// MultiHeadSelfAttention_5669356830878
// MI455X (gfx1250) — matrix operand order repaired and checked by reading, not compiled, not run on hardware
//
#include <hip/hip_runtime.h>
#include <hip/hip_bf16.h>

typedef __attribute__((ext_vector_type(16))) _Float16 v16h;
typedef __attribute__((ext_vector_type(8)))  float    v8f;

#define DIMD   1024
#define NHEAD  16
#define HDIM   64
#define SEQT   2048
#define BATCH  2
#define MROWS  (BATCH * SEQT)   // 4096
#define NQKV   (3 * DIMD)       // 3072

__device__ __forceinline__ v8f wmma16(v16h a, v16h b, v8f c) {
  return __builtin_amdgcn_wmma_f32_16x16x32_f16(
      /*neg_a=*/false, a, /*neg_b=*/false, b,
      /*c_mod=*/(short)0, c, /*reuse_a=*/false, /*reuse_b=*/false);
}

// ---------------------------------------------------------------- convert
__global__ void cvt_f32_to_f16(const float* __restrict__ src,
                               _Float16* __restrict__ dst, int n) {
  int i = blockIdx.x * blockDim.x + threadIdx.x;
  int stride = gridDim.x * blockDim.x;
  for (; i < n; i += stride) dst[i] = (_Float16)src[i];
}

// Transpose-convert: src f32 [rows][cols] row-major -> dst f16 [cols][rows].
__global__ __launch_bounds__(256)
void cvt_transpose_f16(const float* __restrict__ src, _Float16* __restrict__ dst,
                       int rows, int cols) {
  __shared__ float tile[32][33];
  const int bx = blockIdx.x * 32;  // col base
  const int by = blockIdx.y * 32;  // row base
  const int tx = threadIdx.x;      // 0..31
  const int ty = threadIdx.y;      // 0..7
#pragma unroll
  for (int i = 0; i < 32; i += 8)
    tile[ty + i][tx] = src[(size_t)(by + ty + i) * cols + bx + tx];
  __syncthreads();
#pragma unroll
  for (int i = 0; i < 32; i += 8)
    dst[(size_t)(bx + ty + i) * rows + by + tx] = (_Float16)tile[tx][ty + i];
}

// ---------------------------------------------------------------- QKV GEMM
// C[4096,3072] = Xh[4096,1024] @ W + bias, W pre-transposed (WT[n][k]).
// Block tile 128(M) x 64(N); 4 waves, each wave 32 rows x 64 cols.
// sel (q/k/v) and head h are uniform per block since Ntile=64 divides 1024.
//   Q,K : [B*H][T][Hd]   (t-major)
//   V   : [B*H][Hd][T]   (d-major)
__global__ __launch_bounds__(128)
void gemm_qkv_kernel(const _Float16* __restrict__ Xh,
                     const _Float16* __restrict__ WT,   // [3072][1024]
                     const float* __restrict__ bias,
                     _Float16* __restrict__ Qh,
                     _Float16* __restrict__ Kh,
                     _Float16* __restrict__ Vh) {
  constexpr int K = DIMD;
  constexpr int ASTR = 40;   // 80B rows, 16B aligned
  constexpr int BSTR = 40;
  __shared__ _Float16 At[128 * ASTR];  // [m][k]  128x32
  __shared__ _Float16 Bt[64 * BSTR];   // [n][k]   64x32 (transposed tile)

  const int tid = threadIdx.x;
  const int wave = tid >> 5;
  const int lane = tid & 31;
  const int lrow = lane & 15;
  const int lhalf = lane >> 4;
  const int m0 = blockIdx.y * 128;
  const int n0 = blockIdx.x * 64;

  v8f acc[2][4];
#pragma unroll
  for (int g = 0; g < 2; ++g)
#pragma unroll
    for (int nt = 0; nt < 4; ++nt)
#pragma unroll
      for (int r = 0; r < 8; ++r) acc[g][nt][r] = 0.0f;

  for (int k0 = 0; k0 < K; k0 += 32) {
    {
      const _Float16* srcA = Xh + (size_t)(m0 + tid) * K + k0;   // one row/thread
#pragma unroll
      for (int c = 0; c < 32; ++c) At[tid * ASTR + c] = srcA[c];
      int rb = tid >> 1, cb = (tid & 1) * 16;
      const _Float16* srcB = WT + (size_t)(n0 + rb) * K + k0 + cb;
#pragma unroll
      for (int c = 0; c < 16; ++c) Bt[rb * BSTR + cb + c] = srcB[c];
    }
    __syncthreads();

    v16h a[2];
#pragma unroll
    for (int g = 0; g < 2; ++g) {
      const _Float16* ap = &At[(wave * 32 + g * 16 + lrow) * ASTR + lhalf * 16];
#pragma unroll
      for (int e = 0; e < 16; e += 2) {
        int col = e;
        a[g][e] = ap[col];
        a[g][e + 1] = ap[col + 1];
      }
    }
    v16h b[4];
#pragma unroll
    for (int nt = 0; nt < 4; ++nt) {
      const _Float16* bp = &Bt[(nt * 16 + lrow) * BSTR + lhalf * 16];
#pragma unroll
      for (int e = 0; e < 16; ++e) b[nt][e] = bp[e];
    }
#pragma unroll
    for (int nt = 0; nt < 4; ++nt)
#pragma unroll
      for (int g = 0; g < 2; ++g)
        acc[g][nt] = wmma16(a[g], b[nt], acc[g][nt]);
    __syncthreads();
  }

  // epilogue: sel and h are block-uniform
  const int sel = n0 >> 10;
  const int h = (n0 & (DIMD - 1)) >> 6;
  if (sel == 2) {
#pragma unroll
    for (int g = 0; g < 2; ++g)
#pragma unroll
      for (int nt = 0; nt < 4; ++nt)
#pragma unroll
        for (int r = 0; r < 8; ++r) {
          int row = m0 + wave * 32 + g * 16 + r + lhalf * 8;
          int bb = row >> 11, t = row & (SEQT - 1);
          int d = nt * 16 + lrow;
          float v = acc[g][nt][r] + bias[n0 + d];
          Vh[((size_t)(bb * NHEAD + h) * HDIM + d) * SEQT + t] = (_Float16)v;
        }
  } else {
    _Float16* dst = (sel == 0) ? Qh : Kh;
#pragma unroll
    for (int g = 0; g < 2; ++g)
#pragma unroll
      for (int nt = 0; nt < 4; ++nt)
#pragma unroll
        for (int r = 0; r < 8; ++r) {
          int row = m0 + wave * 32 + g * 16 + r + lhalf * 8;
          int bb = row >> 11, t = row & (SEQT - 1);
          int d = nt * 16 + lrow;
          float v = acc[g][nt][r] + bias[n0 + d];
          dst[((size_t)(bb * NHEAD + h) * SEQT + t) * HDIM + d] = (_Float16)v;
        }
  }
}

// ---------------------------------------------------------------- attention
// One block = 4 waves = one 64-row Q tile of one (b,h). Flash-style online
// softmax with causal mask. K tile LDS layout [t][d]; V tile [d][t] so all
// B-fragment reads are contiguous 32B runs. B-frags preloaded in groups so
// WMMAs issue back-to-back under one dscnt wait.
__global__ __launch_bounds__(128)
void attn_kernel(const _Float16* __restrict__ Qh,   // [BH][T][Hd]
                 const _Float16* __restrict__ Kh,   // [BH][T][Hd]
                 const _Float16* __restrict__ Vh,   // [BH][Hd][T]
                 _Float16* __restrict__ atth) {     // [B*T][D]
  constexpr int KSTR = 72;  // 144B rows, 16B aligned
  __shared__ _Float16 Kt[64 * KSTR];   // [k'][d]
  __shared__ _Float16 Vt[64 * KSTR];   // [d][k']  (transposed)
  __shared__ _Float16 Pt[4 * 16 * 64];

  const int qt = blockIdx.x;           // q tile
  const int bh = blockIdx.y;           // b*16+h
  const int tid = threadIdx.x;
  const int wave = tid >> 5;
  const int lane = tid & 31;
  const int lrow = lane & 15;
  const int lhalf = lane >> 4;

  const _Float16* Qb = Qh + (size_t)bh * SEQT * HDIM;
  const _Float16* Kb = Kh + (size_t)bh * SEQT * HDIM;
  const _Float16* Vb = Vh + (size_t)bh * HDIM * SEQT;

  const int qbase = qt * 64 + wave * 16;

  // preload Q A-fragments (16 rows x 64 d, two 16x32 frags)
  v16h qa[2];
#pragma unroll
  for (int dk = 0; dk < 2; ++dk) {
    const _Float16* qp = Qb + (size_t)(qbase + lrow) * HDIM + dk * 32 + lhalf * 16;
#pragma unroll
    for (int e = 0; e < 16; e += 2) {
      int col = e;
      qa[dk][e] = qp[col];
      qa[dk][e + 1] = qp[col + 1];
    }
  }

  float mrun[8], lsum[8];
  v8f oacc[4];
#pragma unroll
  for (int r = 0; r < 8; ++r) { mrun[r] = -3.0e38f; lsum[r] = 0.0f; }
#pragma unroll
  for (int dt = 0; dt < 4; ++dt)
#pragma unroll
    for (int r = 0; r < 8; ++r) oacc[dt][r] = 0.0f;

  _Float16* pw = &Pt[wave * 16 * 64];

  for (int kt = 0; kt <= qt; ++kt) {
    // stage K tile [k'][d] and V tile [d][k']; both contiguous both sides
    {
      int r = tid >> 1, c0 = (tid & 1) * 32;
      const _Float16* ks = Kb + (size_t)(kt * 64 + r) * HDIM + c0;     // row t
      const _Float16* vs = Vb + (size_t)r * SEQT + kt * 64 + c0;       // row d
#pragma unroll
      for (int c = 0; c < 32; ++c) {
        Kt[r * KSTR + c0 + c] = ks[c];
        Vt[r * KSTR + c0 + c] = vs[c];
      }
    }
    __syncthreads();

    // S = Q K^T : preload all 8 B-frags, then 8 WMMAs back-to-back
    v16h bk[8];
#pragma unroll
    for (int nt = 0; nt < 4; ++nt)
#pragma unroll
      for (int dk = 0; dk < 2; ++dk) {
        const _Float16* kp = &Kt[(nt * 16 + lrow) * KSTR + dk * 32 + lhalf * 16];
#pragma unroll
        for (int e = 0; e < 16; ++e) bk[nt * 2 + dk][e] = kp[e];
      }
    v8f s[4];
#pragma unroll
    for (int nt = 0; nt < 4; ++nt) {
#pragma unroll
      for (int r = 0; r < 8; ++r) s[nt][r] = 0.0f;
      s[nt] = wmma16(qa[0], bk[nt * 2 + 0], s[nt]);
      s[nt] = wmma16(qa[1], bk[nt * 2 + 1], s[nt]);
    }

    // scale + causal mask + row max
    float rmax[8];
#pragma unroll
    for (int r = 0; r < 8; ++r) rmax[r] = -3.0e38f;
#pragma unroll
    for (int nt = 0; nt < 4; ++nt) {
      int coln = kt * 64 + nt * 16 + lrow;
#pragma unroll
      for (int r = 0; r < 8; ++r) {
        int rowq = qbase + r + lhalf * 8;
        float v = s[nt][r] * 0.125f;             // 1/sqrt(64)
        v = (coln <= rowq) ? v : -3.0e38f;
        s[nt][r] = v;
        rmax[r] = fmaxf(rmax[r], v);
      }
    }
#pragma unroll
    for (int msk = 1; msk < 16; msk <<= 1)
#pragma unroll
      for (int r = 0; r < 8; ++r)
        rmax[r] = fmaxf(rmax[r], __shfl_xor(rmax[r], msk, 32));

    // online softmax update
    float corr[8], mnew[8], rsum[8];
#pragma unroll
    for (int r = 0; r < 8; ++r) {
      mnew[r] = fmaxf(mrun[r], rmax[r]);
      corr[r] = __expf(mrun[r] - mnew[r]);
      mrun[r] = mnew[r];
      rsum[r] = 0.0f;
    }
#pragma unroll
    for (int nt = 0; nt < 4; ++nt)
#pragma unroll
      for (int r = 0; r < 8; ++r) {
        float p = __expf(s[nt][r] - mnew[r]);
        rsum[r] += p;
        pw[(r + lhalf * 8) * 64 + nt * 16 + lrow] = (_Float16)p;
      }
#pragma unroll
    for (int msk = 1; msk < 16; msk <<= 1)
#pragma unroll
      for (int r = 0; r < 8; ++r)
        rsum[r] += __shfl_xor(rsum[r], msk, 32);
#pragma unroll
    for (int r = 0; r < 8; ++r) lsum[r] = lsum[r] * corr[r] + rsum[r];
#pragma unroll
    for (int dt = 0; dt < 4; ++dt)
#pragma unroll
      for (int r = 0; r < 8; ++r) oacc[dt][r] *= corr[r];

    // O += P @ V : per k-chunk preload A + all 4 V B-frags, then 4 WMMAs
#pragma unroll
    for (int kk = 0; kk < 2; ++kk) {
      v16h pa;
      const _Float16* pp = pw + lrow * 64 + kk * 32 + lhalf * 16;
#pragma unroll
      for (int e = 0; e < 16; e += 2) {
        int col = e;
        pa[e] = pp[col];
        pa[e + 1] = pp[col + 1];
      }
      v16h vb[4];
#pragma unroll
      for (int dt = 0; dt < 4; ++dt) {
        const _Float16* vp = &Vt[(dt * 16 + lrow) * KSTR + kk * 32 + lhalf * 16];
#pragma unroll
        for (int e = 0; e < 16; ++e) vb[dt][e] = vp[e];
      }
#pragma unroll
      for (int dt = 0; dt < 4; ++dt)
        oacc[dt] = wmma16(pa, vb[dt], oacc[dt]);
    }
    __syncthreads();
  }

  // normalize and write to att [B*T][D] f16
  const int bb = bh >> 4, h = bh & 15;
#pragma unroll
  for (int dt = 0; dt < 4; ++dt)
#pragma unroll
    for (int r = 0; r < 8; ++r) {
      int t = qbase + r + lhalf * 8;
      float v = oacc[dt][r] / lsum[r];
      atth[((size_t)(bb * SEQT + t)) * DIMD + h * 64 + dt * 16 + lrow] = (_Float16)v;
    }
}

// ---------------------------------------------------------------- out proj
// out[4096,1024] = att[4096,1024] @ Wout + b_out, Wout pre-transposed.
// Block tile 128(M) x 64(N).
__global__ __launch_bounds__(128)
void gemm_out_kernel(const _Float16* __restrict__ Ah,
                     const _Float16* __restrict__ WT,   // [1024][1024] (n-major)
                     const float* __restrict__ bias,
                     float* __restrict__ out) {
  constexpr int K = DIMD, N = DIMD;
  constexpr int ASTR = 40, BSTR = 40;
  __shared__ _Float16 At[128 * ASTR];
  __shared__ _Float16 Bt[64 * BSTR];

  const int tid = threadIdx.x;
  const int wave = tid >> 5;
  const int lane = tid & 31;
  const int lrow = lane & 15;
  const int lhalf = lane >> 4;
  const int m0 = blockIdx.y * 128;
  const int n0 = blockIdx.x * 64;

  v8f acc[2][4];
#pragma unroll
  for (int g = 0; g < 2; ++g)
#pragma unroll
    for (int nt = 0; nt < 4; ++nt)
#pragma unroll
      for (int r = 0; r < 8; ++r) acc[g][nt][r] = 0.0f;

  for (int k0 = 0; k0 < K; k0 += 32) {
    {
      const _Float16* srcA = Ah + (size_t)(m0 + tid) * K + k0;
#pragma unroll
      for (int c = 0; c < 32; ++c) At[tid * ASTR + c] = srcA[c];
      int rb = tid >> 1, cb = (tid & 1) * 16;
      const _Float16* srcB = WT + (size_t)(n0 + rb) * K + k0 + cb;
#pragma unroll
      for (int c = 0; c < 16; ++c) Bt[rb * BSTR + cb + c] = srcB[c];
    }
    __syncthreads();

    v16h a[2];
#pragma unroll
    for (int g = 0; g < 2; ++g) {
      const _Float16* ap = &At[(wave * 32 + g * 16 + lrow) * ASTR + lhalf * 16];
#pragma unroll
      for (int e = 0; e < 16; e += 2) {
      int col = e;
        a[g][e] = ap[col];
        a[g][e + 1] = ap[col + 1];
      }
    }
    v16h b[4];
#pragma unroll
    for (int nt = 0; nt < 4; ++nt) {
      const _Float16* bp = &Bt[(nt * 16 + lrow) * BSTR + lhalf * 16];
#pragma unroll
      for (int e = 0; e < 16; ++e) b[nt][e] = bp[e];
    }
#pragma unroll
    for (int nt = 0; nt < 4; ++nt)
#pragma unroll
      for (int g = 0; g < 2; ++g)
        acc[g][nt] = wmma16(a[g], b[nt], acc[g][nt]);
    __syncthreads();
  }

#pragma unroll
  for (int g = 0; g < 2; ++g)
#pragma unroll
    for (int nt = 0; nt < 4; ++nt)
#pragma unroll
      for (int r = 0; r < 8; ++r) {
        int row = m0 + wave * 32 + g * 16 + r + lhalf * 8;
        int col = n0 + nt * 16 + lrow;
        out[(size_t)row * N + col] = acc[g][nt][r] + bias[col];
      }
}

// ---------------------------------------------------------------- launch
extern "C" void kernel_launch(void* const* d_in, const int* in_sizes, int n_in,
                              void* d_out, int out_size, void* d_ws, size_t ws_size,
                              hipStream_t stream) {
  const float* x     = (const float*)d_in[0];
  const float* W_qkv = (const float*)d_in[1];
  const float* b_qkv = (const float*)d_in[2];
  const float* W_out = (const float*)d_in[3];
  const float* b_out = (const float*)d_in[4];
  float* out = (float*)d_out;

  char* ws = (char*)d_ws;
  size_t off = 0;
  auto carve = [&](size_t bytes) -> void* {
    void* p = ws + off;
    off += (bytes + 255) & ~(size_t)255;
    return p;
  };
  _Float16* xh     = (_Float16*)carve((size_t)MROWS * DIMD * 2);
  _Float16* wqkvT  = (_Float16*)carve((size_t)DIMD * NQKV * 2);  // [3072][1024]
  _Float16* woutT  = (_Float16*)carve((size_t)DIMD * DIMD * 2);  // [1024][1024]
  _Float16* Qh     = (_Float16*)carve((size_t)MROWS * DIMD * 2);
  _Float16* Kh     = (_Float16*)carve((size_t)MROWS * DIMD * 2);
  _Float16* Vh     = (_Float16*)carve((size_t)MROWS * DIMD * 2);
  _Float16* atth   = (_Float16*)carve((size_t)MROWS * DIMD * 2);

  cvt_f32_to_f16<<<1024, 256, 0, stream>>>(x, xh, MROWS * DIMD);
  cvt_transpose_f16<<<dim3(NQKV / 32, DIMD / 32), dim3(32, 8), 0, stream>>>(
      W_qkv, wqkvT, DIMD, NQKV);
  cvt_transpose_f16<<<dim3(DIMD / 32, DIMD / 32), dim3(32, 8), 0, stream>>>(
      W_out, woutT, DIMD, DIMD);

  gemm_qkv_kernel<<<dim3(NQKV / 64, MROWS / 128), 128, 0, stream>>>(
      xh, wqkvT, b_qkv, Qh, Kh, Vh);

  attn_kernel<<<dim3(SEQT / 64, BATCH * NHEAD), 128, 0, stream>>>(
      Qh, Kh, Vh, atth);

  gemm_out_kernel<<<dim3(DIMD / 64, MROWS / 128), 128, 0, stream>>>(
      atth, woutT, b_out, out);
}
